// RoPESelfAttention_57294863729080
// MI455X (gfx1250) — matrix operand order repaired and checked by reading, not compiled, not run on hardware
//
#include <hip/hip_runtime.h>
#include <hip/hip_bf16.h>

// ---------------------------------------------------------------------------
// RoPE causal self-attention, MI455X (gfx1250, wave32, WMMA bf16 + TDM)
//   B=2, T=2048, C=2048, H=16, D=128
// Pipeline (all on `stream`):
//   1) cvt_bf16   : x fp32 -> bf16
//   2) cvtT  x4   : W fp32 -> bf16 TRANSPOSED (so GEMM B operand is K-contig)
//   3) gemm  x3   : Q/K/V = xb @ W  (v_wmma_f32_16x16x32_bf16, f32 accum)
//                   fused RoPE epilogue (Q,K) / permute (V) -> [B,H,T,D] bf16
//   4) attn       : flash attention per (b,h,128-query tile); K/V tiles pulled
//                   Global->LDS by the Tensor Data Mover (tensor_load_to_lds,
//                   s_wait_tensorcnt), QK^T and PV on WMMA, online softmax
//   5) gemm mode0 : out = O @ Wo -> d_out fp32
// Workspace layout (112 MiB):
//   [0,16M)  xb   [16M..48M) WqT/WkT/WvT/WoT  [48M..96M) Qb/Kb/Vb  [96M..112M) Ob
// ---------------------------------------------------------------------------

typedef __attribute__((ext_vector_type(16))) __bf16       v16bf;
typedef __attribute__((ext_vector_type(8)))  float        v8f;
typedef __attribute__((ext_vector_type(4)))  unsigned int v4u;
typedef __attribute__((ext_vector_type(4)))  float        v4f;
typedef __attribute__((ext_vector_type(8)))  int          i32x8;
typedef __attribute__((ext_vector_type(4)))  int          i32x4;

union Frag  { v4u u4[2]; v16bf v; };
union Pack8 { v4u u4; __bf16 h[8]; };

#if defined(__gfx1250__) && __has_builtin(__builtin_amdgcn_tensor_load_to_lds) && \
    __has_builtin(__builtin_amdgcn_s_wait_tensorcnt)
#define USE_TDM 1
#else
#define USE_TDM 0
#endif

__device__ inline __bf16 f2bf(float f) {
  union { float f; unsigned int u; } a; a.f = f;
  unsigned int u = a.u;
  u += 0x7fffu + ((u >> 16) & 1u);            // round-to-nearest-even
  union { unsigned short s; __bf16 h; } b; b.s = (unsigned short)(u >> 16);
  return b.h;
}

#if USE_TDM
// ---------------------------------------------------------------------------
// Issue a 2-D TDM tile load Global->LDS (Tensor DMA Descriptor per ISA ch.8).
// Element size is forced to 8 bytes for peak DMA width; dims/strides are in
// 8-byte units. group0: count=1 | lds_addr | global_addr[56:0] | type=2.
// ---------------------------------------------------------------------------
__device__ inline void tdm_load_tile_2d(const void* gptr, unsigned lds_off,
                                        int tileRows, int rowUnits8,
                                        int tensorRows) {
  unsigned long long ga = (unsigned long long)(size_t)gptr;
  v4u g0;
  g0[0] = 1u;                                              // count=1 (valid)
  g0[1] = lds_off;                                         // lds_addr (bytes)
  g0[2] = (unsigned)(ga & 0xffffffffu);                    // global_addr lo
  g0[3] = (unsigned)((ga >> 32) & 0x01ffffffu) | 0x80000000u; // hi | type=2
  i32x8 g1;
  g1[0] = (3 << 16);                       // wg_mask=0 | data_size=3 (8B)
  g1[1] = (rowUnits8 & 0xffff) << 16;      // tensor_dim0[15:0]  @ bits 79:48
  g1[2] = (tensorRows & 0xffff) << 16;     // tensor_dim1[15:0]  @ bits 111:80
  g1[3] = (rowUnits8 & 0xffff) << 16;      // tile_dim0          @ bits 127:112
  g1[4] = (tileRows & 0xffff);             // tile_dim1 (tile_dim2 = 0 -> 2D)
  g1[5] = rowUnits8;                       // tensor_dim0_stride[31:0]
  g1[6] = 0;                               // stride hi | dim1_stride lo
  g1[7] = 0;
  i32x4 z4 = {0, 0, 0, 0};
#if defined(__clang_major__) && (__clang_major__ >= 23)
  i32x8 z8 = {0, 0, 0, 0, 0, 0, 0, 0};
  __builtin_amdgcn_tensor_load_to_lds(g0, g1, z4, z4, z8, 0);
#else
  __builtin_amdgcn_tensor_load_to_lds(g0, g1, z4, z4, 0);
#endif
}
#endif

// ---------------------------------------------------------------------------
// fp32 -> bf16 straight conversion (4 elems/thread, vectorized)
// ---------------------------------------------------------------------------
__global__ __launch_bounds__(256) void cvt_bf16(const float* __restrict__ s,
                                                __bf16* __restrict__ d, int n) {
  int i = (blockIdx.x * 256 + threadIdx.x) * 4;
  if (i + 3 < n) {
    v4f f = *(const v4f*)(s + i);
    union { __bf16 h[4]; unsigned long long u; } p;
#pragma unroll
    for (int e = 0; e < 4; ++e) p.h[e] = f2bf(f[e]);
    *(unsigned long long*)(d + i) = p.u;
  }
}

// ---------------------------------------------------------------------------
// fp32 [2048,2048] -> bf16 transposed, 32x32 LDS tile transpose
// ---------------------------------------------------------------------------
__global__ __launch_bounds__(256) void cvtT(const float* __restrict__ W,
                                            __bf16* __restrict__ WT) {
  __shared__ float tile[32][33];
  const int bx = blockIdx.x * 32, by = blockIdx.y * 32;
  const int tx = threadIdx.x & 31, ty = threadIdx.x >> 5;
#pragma unroll
  for (int j = ty; j < 32; j += 8) tile[j][tx] = W[(size_t)(by + j) * 2048 + bx + tx];
  __syncthreads();
#pragma unroll
  for (int j = ty; j < 32; j += 8)
    WT[(size_t)(bx + j) * 2048 + by + tx] = f2bf(tile[tx][j]);
}

// ---------------------------------------------------------------------------
// bf16 WMMA GEMM: out[M,2048] = A[M,K] @ Bt[2048,K]^T   (Bt is W transposed)
// 128x128 block tile, 8 waves (4x2), 32-deep K steps, f32 accumulation.
// mode 0: store fp32 to outF[M,2048]
// mode 1/2: RoPE epilogue (Q/K) -> outP bf16 [B,H,T,D]
// mode 3: permute only (V)      -> outP bf16 [B,H,T,D]
// ---------------------------------------------------------------------------
__global__ __launch_bounds__(256) void gemm_bf16_wmma(
    const __bf16* __restrict__ A, const __bf16* __restrict__ Bt,
    float* __restrict__ outF, __bf16* __restrict__ outP, int mode, int Kdim) {
  extern __shared__ __align__(16) char smem_raw[];      // 64 KB
  float*  smem = (float*)smem_raw;                      // epilogue 128x128 f32
  __bf16* As   = (__bf16*)smem_raw;                     // 128x32
  __bf16* Bs   = As + 128 * 32;                         // 128x32
  const int tid = threadIdx.x, lane = tid & 31, w = tid >> 5;
  const int colh = lane & 15, hi = lane >> 4;
  const int n0 = blockIdx.x * 128, m0 = blockIdx.y * 128;
  const int wm = (w & 3) * 32;            // wave row base (2 m-tiles)
  const int wn = (w >> 2) * 64;           // wave col base (4 n-tiles)

  v8f acc[2][4];
#pragma unroll
  for (int mt = 0; mt < 2; ++mt)
#pragma unroll
    for (int nt = 0; nt < 4; ++nt)
#pragma unroll
      for (int e = 0; e < 8; ++e) acc[mt][nt][e] = 0.f;

  const v4u* A4 = (const v4u*)A;
  const v4u* B4 = (const v4u*)Bt;
  v4u* As4 = (v4u*)As;
  v4u* Bs4 = (v4u*)Bs;

  for (int k0 = 0; k0 < Kdim; k0 += 32) {
#pragma unroll
    for (int i = 0; i < 2; ++i) {                 // 512 v4u each for A and B
      int idx = tid + i * 256;
      int rr = idx >> 2, cc = idx & 3;
      As4[idx] = A4[(((size_t)(m0 + rr) * Kdim + k0) >> 3) + cc];
      Bs4[idx] = B4[(((size_t)(n0 + rr) * Kdim + k0) >> 3) + cc];
    }
    __syncthreads();

    Frag aF[2], bF[4];
#pragma unroll
    for (int mt = 0; mt < 2; ++mt) {              // A: K-contig per lane
      const __bf16* p = As + (wm + mt * 16 + colh) * 32 + hi * 16;
      aF[mt].u4[0] = *(const v4u*)p;
      aF[mt].u4[1] = *(const v4u*)(p + 8);
    }
#pragma unroll
    for (int nt = 0; nt < 4; ++nt) {              // B: K-contig per lane
      const __bf16* p = Bs + (wn + nt * 16 + colh) * 32 + hi * 16;
      bF[nt].u4[0] = *(const v4u*)p;
      bF[nt].u4[1] = *(const v4u*)(p + 8);
    }
#pragma unroll
    for (int mt = 0; mt < 2; ++mt)
#pragma unroll
      for (int nt = 0; nt < 4; ++nt)
        acc[mt][nt] = __builtin_amdgcn_wmma_f32_16x16x32_bf16(
            false, aF[mt].v, false, bF[nt].v, (short)0, acc[mt][nt], false, false);
    __syncthreads();
  }

  if (mode == 0) {                                // plain fp32 store
#pragma unroll
    for (int mt = 0; mt < 2; ++mt)
#pragma unroll
      for (int nt = 0; nt < 4; ++nt)
#pragma unroll
        for (int r = 0; r < 8; ++r) {
          int gm = m0 + wm + mt * 16 + r + 8 * hi;
          int gn = n0 + wn + nt * 16 + colh;
          outF[(size_t)gm * 2048 + gn] = acc[mt][nt][r];
        }
    return;
  }

  // RoPE / permute epilogue: block col tile == exactly one head (128 = D),
  // exchange rotate-half partner (d ^ 64) through the LDS f32 tile.
#pragma unroll
  for (int mt = 0; mt < 2; ++mt)
#pragma unroll
    for (int nt = 0; nt < 4; ++nt)
#pragma unroll
      for (int r = 0; r < 8; ++r)
        smem[(wm + mt * 16 + r + 8 * hi) * 128 + wn + nt * 16 + colh] =
            acc[mt][nt][r];
  __syncthreads();

  const int  h    = n0 >> 7;
  const bool rope = (mode != 3);
  for (int i = 0; i < 64; ++i) {
    int idx = i * 256 + tid;
    int m = idx >> 7, d = idx & 127;
    int gm = m0 + m;
    int b = gm >> 11, t = gm & 2047;
    float u = smem[m * 128 + d];
    float o = u;
    if (rope) {
      float up  = smem[m * 128 + (d ^ 64)];
      float inv = __powf(10000.f, -(float)(d & 63) * (1.0f / 64.0f));
      float sv, cv;
      __sincosf((float)t * inv, &sv, &cv);
      o = (d < 64) ? (u * cv - up * sv) : (u * cv + up * sv);
    }
    outP[(((size_t)b * 16 + h) * 2048 + t) * 128 + d] = f2bf(o);
  }
}

// ---------------------------------------------------------------------------
// Flash attention, causal. One block = (b, h, 128 query rows); 8 waves, each
// wave owns 16 query rows. LDS: Ks[key][dim] 32K + Vs[dim][key] 32K + Ps 32K
// (C-layout -> A-layout re-shuffle) + Vr 32K (TDM row-major V staging).
// K/V tiles are moved Global->LDS by the Tensor Data Mover (wave 0 issues the
// descriptors, waits on TENSORcnt, then the workgroup barrier publishes).
// ---------------------------------------------------------------------------
__global__ __launch_bounds__(256) void attn_wmma(
    const __bf16* __restrict__ Q, const __bf16* __restrict__ K,
    const __bf16* __restrict__ V, __bf16* __restrict__ O) {
  extern __shared__ __align__(16) char smem_raw[];
  __bf16* Ks = (__bf16*)smem_raw;      // 128x128 [key][dim]
  __bf16* Vs = Ks + 128 * 128;         // 128x128 [dim][key]
  __bf16* Ps = Vs + 128 * 128;         // 8 waves x 16 rows x 128 keys
  __bf16* Vr = Ps + 128 * 128;         // 128x128 [key][dim] (TDM staging)
  const int tid = threadIdx.x, lane = tid & 31, w = tid >> 5;
  const int colh = lane & 15, hi = lane >> 4;
  const int qb = blockIdx.x, h = blockIdx.y, b = blockIdx.z;
  const size_t headOff = ((size_t)b * 16 + h) * 2048 * 128;
  const __bf16* Qh = Q + headOff;
  const __bf16* Kh = K + headOff;
  const __bf16* Vh = V + headOff;

  // Persistent Q fragments for this wave's 16 rows (4 K-steps of 32 dims).
  Frag qF[4];
  {
    const __bf16* qp = Qh + (size_t)(qb * 128 + w * 16 + colh) * 128 + hi * 16;
#pragma unroll
    for (int ks = 0; ks < 4; ++ks) {
      qF[ks].u4[0] = *(const v4u*)(qp + ks * 32);
      qF[ks].u4[1] = *(const v4u*)(qp + ks * 32 + 8);
    }
  }

  v8f   oacc[8];
  float mrow[8], lrow[8];
#pragma unroll
  for (int dt = 0; dt < 8; ++dt)
#pragma unroll
    for (int e = 0; e < 8; ++e) oacc[dt][e] = 0.f;
#pragma unroll
  for (int r = 0; r < 8; ++r) { mrow[r] = -1e30f; lrow[r] = 0.f; }

  const float scale = 0.08838834764831845f;   // 1/sqrt(128)

  for (int jb = 0; jb <= qb; ++jb) {
#if USE_TDM
    if (w == 0) {                               // TDM: async DMA Global->LDS
      // 128 rows x 256B (= 32 x 8B units) per tile.
      tdm_load_tile_2d(Kh + (size_t)jb * 128 * 128,
                       (unsigned)(size_t)Ks, 128, 32, 2048);
      tdm_load_tile_2d(Vh + (size_t)jb * 128 * 128,
                       (unsigned)(size_t)Vr, 128, 32, 2048);
      __builtin_amdgcn_s_wait_tensorcnt(0);
    }
    __syncthreads();
#pragma unroll
    for (int i = 0; i < 8; ++i) {               // LDS->LDS transpose Vr -> Vs
      int idx = i * 256 + tid;
      int key = idx >> 4, c = (idx & 15) * 8;
      Pack8 pk;
      pk.u4 = *(const v4u*)(Vr + key * 128 + c);
#pragma unroll
      for (int e = 0; e < 8; ++e) Vs[(c + e) * 128 + key] = pk.h[e];
    }
    __syncthreads();
#else
#pragma unroll
    for (int i = 0; i < 8; ++i) {               // K tile -> LDS (row-major)
      int idx = i * 256 + tid;
      int key = idx >> 4, c = (idx & 15) * 8;
      ((v4u*)Ks)[idx] = *(const v4u*)(Kh + (size_t)(jb * 128 + key) * 128 + c);
    }
#pragma unroll
    for (int i = 0; i < 8; ++i) {               // V tile -> LDS transposed
      int idx = i * 256 + tid;
      int key = idx >> 4, c = (idx & 15) * 8;
      Pack8 pk;
      pk.u4 = *(const v4u*)(Vh + (size_t)(jb * 128 + key) * 128 + c);
#pragma unroll
      for (int e = 0; e < 8; ++e) Vs[(c + e) * 128 + key] = pk.h[e];
    }
    __syncthreads();
#endif

    // S = Q K^T : 8 key-tiles x 4 K-steps of WMMA
    v8f st[8];
#pragma unroll
    for (int nt = 0; nt < 8; ++nt)
#pragma unroll
      for (int e = 0; e < 8; ++e) st[nt][e] = 0.f;
#pragma unroll
    for (int ks = 0; ks < 4; ++ks)
#pragma unroll
      for (int nt = 0; nt < 8; ++nt) {
        Frag kf;
        const __bf16* p = Ks + (nt * 16 + colh) * 128 + ks * 32 + hi * 16;
        kf.u4[0] = *(const v4u*)p;
        kf.u4[1] = *(const v4u*)(p + 8);
        st[nt] = __builtin_amdgcn_wmma_f32_16x16x32_bf16(
            false, qF[ks].v, false, kf.v, (short)0, st[nt], false, false);
      }

    // Online softmax (rows live on 16-lane half-waves -> shfl_xor reduce).
    const bool diag = (jb == qb);
#pragma unroll
    for (int r = 0; r < 8; ++r) {
      int qloc = w * 16 + r + 8 * hi;
      float mx = -1e30f;
#pragma unroll
      for (int nt = 0; nt < 8; ++nt) {
        float sv = st[nt][r] * scale;
        if (diag && (nt * 16 + colh) > qloc) sv = -1e30f;
        st[nt][r] = sv;
        mx = fmaxf(mx, sv);
      }
#pragma unroll
      for (int off = 1; off < 16; off <<= 1) mx = fmaxf(mx, __shfl_xor(mx, off, 32));
      float mn = fmaxf(mrow[r], mx);
      float ssum = 0.f;
#pragma unroll
      for (int nt = 0; nt < 8; ++nt) {
        float p = __expf(st[nt][r] - mn);
        st[nt][r] = p;
        ssum += p;
      }
#pragma unroll
      for (int off = 1; off < 16; off <<= 1) ssum += __shfl_xor(ssum, off, 32);
      float alpha = __expf(mrow[r] - mn);
      lrow[r] = lrow[r] * alpha + ssum;
      mrow[r] = mn;
#pragma unroll
      for (int dt = 0; dt < 8; ++dt) oacc[dt][r] *= alpha;
    }

    // P (C-layout f32) -> per-wave LDS patch as bf16, reload as A-fragments.
    __bf16* Pw = Ps + w * 16 * 128;
#pragma unroll
    for (int nt = 0; nt < 8; ++nt)
#pragma unroll
      for (int r = 0; r < 8; ++r)
        Pw[(r + 8 * hi) * 128 + nt * 16 + colh] = f2bf(st[nt][r]);

    // O += P @ V
#pragma unroll
    for (int ks = 0; ks < 4; ++ks) {
      Frag pf;
      const __bf16* pp = Pw + colh * 128 + ks * 32 + hi * 16;
      pf.u4[0] = *(const v4u*)pp;
      pf.u4[1] = *(const v4u*)(pp + 8);
#pragma unroll
      for (int dt = 0; dt < 8; ++dt) {
        Frag vf;
        const __bf16* vp = Vs + (dt * 16 + colh) * 128 + ks * 32 + hi * 16;
        vf.u4[0] = *(const v4u*)vp;
        vf.u4[1] = *(const v4u*)(vp + 8);
        oacc[dt] = __builtin_amdgcn_wmma_f32_16x16x32_bf16(
            false, pf.v, false, vf.v, (short)0, oacc[dt], false, false);
      }
    }
    __syncthreads();
  }

  // Normalize, convert, store to O[b*T + row][h*128 + dim] (bf16, row-major)
#pragma unroll
  for (int r = 0; r < 8; ++r) {
    float inv = 1.0f / lrow[r];
    int row = qb * 128 + w * 16 + r + 8 * hi;
    size_t base = ((size_t)b * 2048 + row) * 2048 + h * 128;
#pragma unroll
    for (int dt = 0; dt < 8; ++dt)
      O[base + dt * 16 + colh] = f2bf(oacc[dt][r] * inv);
  }
}

// ---------------------------------------------------------------------------
extern "C" void kernel_launch(void* const* d_in, const int* in_sizes, int n_in,
                              void* d_out, int out_size, void* d_ws, size_t ws_size,
                              hipStream_t stream) {
  (void)in_sizes; (void)n_in; (void)out_size; (void)ws_size;
  const float* x  = (const float*)d_in[0];
  const float* Wq = (const float*)d_in[1];
  const float* Wk = (const float*)d_in[2];
  const float* Wv = (const float*)d_in[3];
  const float* Wo = (const float*)d_in[4];

  char* ws = (char*)d_ws;
  const size_t MB = 1024u * 1024u;
  __bf16* xb  = (__bf16*)(ws + 0);
  __bf16* WqT = (__bf16*)(ws + 16 * MB);
  __bf16* WkT = (__bf16*)(ws + 24 * MB);
  __bf16* WvT = (__bf16*)(ws + 32 * MB);
  __bf16* WoT = (__bf16*)(ws + 40 * MB);
  __bf16* Qb  = (__bf16*)(ws + 48 * MB);
  __bf16* Kb  = (__bf16*)(ws + 64 * MB);
  __bf16* Vb  = (__bf16*)(ws + 80 * MB);
  __bf16* Ob  = (__bf16*)(ws + 96 * MB);

  cvt_bf16<<<8192, 256, 0, stream>>>(x, xb, 4096 * 2048);
  dim3 tg(64, 64);
  cvtT<<<tg, 256, 0, stream>>>(Wq, WqT);
  cvtT<<<tg, 256, 0, stream>>>(Wk, WkT);
  cvtT<<<tg, 256, 0, stream>>>(Wv, WvT);
  cvtT<<<tg, 256, 0, stream>>>(Wo, WoT);

  dim3 gg(16, 32);                         // (cols/128, rows/128)
  size_t gemmLds = 128 * 128 * sizeof(float);   // 64 KB
  gemm_bf16_wmma<<<gg, 256, gemmLds, stream>>>(xb, WqT, nullptr, Qb, 1, 2048);
  gemm_bf16_wmma<<<gg, 256, gemmLds, stream>>>(xb, WkT, nullptr, Kb, 2, 2048);
  gemm_bf16_wmma<<<gg, 256, gemmLds, stream>>>(xb, WvT, nullptr, Vb, 3, 2048);

  dim3 ga(16, 16, 2);                      // (T/128, H, B)
  size_t attnLds = 4u * 128 * 128 * sizeof(__bf16);  // 128 KB (<= 320KB/WGP)
  attn_wmma<<<ga, 256, attnLds, stream>>>(Qb, Kb, Vb, Ob);

  gemm_bf16_wmma<<<gg, 256, gemmLds, stream>>>(Ob, WoT, (float*)d_out, nullptr, 0, 2048);
}
